// FancyViT_58755152609780
// MI455X (gfx1250) — compile-verified, matrix operand order checked by reading, not run on hardware
//
#include <hip/hip_runtime.h>

// ---------------- constants (SAM ViT-B window attention) ----------------
#define WSZ    14
#define NHEADS 12
#define HD     64
#define DIMC   768
#define NTOK   196              // 14*14
#define BW     200              // 8 * 5 * 5 windows
#define BH     2400             // BW * NHEADS
#define ROWS   39200            // BW * NTOK
#define QKVC   2304

typedef __attribute__((ext_vector_type(16))) __bf16 bf16x16;
typedef __attribute__((ext_vector_type(8)))  float  f32x8;

union FragBF { bf16x16 v; unsigned short u[16]; };
union FragF  { f32x8   v; float          f[8];  };

__device__ __forceinline__ unsigned short f2bf(float f) {
  unsigned int u = __float_as_uint(f);
  unsigned int r = (u + 0x7FFFu + ((u >> 16) & 1u)) >> 16;   // RNE
  return (unsigned short)r;
}
__device__ __forceinline__ float bf2f(unsigned short b) {
  return __uint_as_float(((unsigned int)b) << 16);
}

// =====================================================================
// Kernel 0a: fused window-partition + pad + f32->bf16 pack of x
//   xwin[R, c], R = window-row (39200), zero rows for padded area
// =====================================================================
__global__ __launch_bounds__(256) void pack_xwin_kernel(
    const float* __restrict__ x, unsigned short* __restrict__ xwin)
{
  int R = blockIdx.x;
  int w = R / NTOK, t = R - w * NTOK;
  int b = w / 25, wrem = w - b * 25;
  int wi = wrem / 5, wj = wrem - wi * 5;
  int ty = t / WSZ, tx = t - ty * WSZ;
  int y = wi * WSZ + ty, xx = wj * WSZ + tx;
  bool valid = (y < 64 && xx < 64);
  const float* src = x + ((size_t)(b * 64 + y) * 64 + xx) * DIMC;
  unsigned short* dst = xwin + (size_t)R * DIMC;
  for (int c = threadIdx.x; c < DIMC; c += 256)
    dst[c] = valid ? f2bf(src[c]) : (unsigned short)0;
}

// Kernel 0b: elementwise f32->bf16 pack (weights)
__global__ __launch_bounds__(256) void pack_w_kernel(
    const float* __restrict__ src, unsigned short* __restrict__ dst, int n)
{
  int i = blockIdx.x * 256 + threadIdx.x;
  if (i < n) dst[i] = f2bf(src[i]);
}

// =====================================================================
// Kernel 1: QKV GEMM (bf16 WMMA, f32 accum), pure bf16 copies into LDS
//   out[R, J] = sum_c xwin[R, c] * wq[J, c] + qkv_b[J]
//   q is pre-scaled by hd^-0.5 at store; scattered to q/k/v buffers
//   laid out [BW*NHEADS][196][64] bf16.
// =====================================================================
__global__ __launch_bounds__(256) void qkv_gemm_kernel(
    const unsigned short* __restrict__ xwin, const unsigned short* __restrict__ wq,
    const float* __restrict__ qkv_b,
    unsigned short* __restrict__ qout, unsigned short* __restrict__ kout,
    unsigned short* __restrict__ vout)
{
  __shared__ __align__(16) unsigned short Alds[128 * 64];
  __shared__ __align__(16) unsigned short Blds[64 * 64];

  const int tid   = threadIdx.x;
  const int lane  = tid & 31;
  const int wave  = tid >> 5;
  const int mbase = blockIdx.y * 128;
  const int nbase = blockIdx.x * 64;
  const int c0    = (lane >> 4) * 8;
  const int l15   = lane & 15;

  FragF acc[4];
  for (int nt = 0; nt < 4; ++nt)
    for (int r = 0; r < 8; ++r) acc[nt].f[r] = 0.0f;

  for (int ks = 0; ks < DIMC; ks += 64) {
    // A tile: 128 rows x 64 cols, vectorized bf16 copies
    for (int i = tid; i < 1024; i += 256) {          // 1024 uint4 chunks
      int r = i >> 3, cg = (i & 7) * 8;
      int R = mbase + r;
      uint4 val = make_uint4(0u, 0u, 0u, 0u);
      if (R < ROWS)
        val = *reinterpret_cast<const uint4*>(xwin + (size_t)R * DIMC + ks + cg);
      *reinterpret_cast<uint4*>(&Alds[r * 64 + cg]) = val;
    }
    // B tile: 64 weight rows x 64 cols
    for (int i = tid; i < 512; i += 256) {
      int r = i >> 3, cg = (i & 7) * 8;
      *reinterpret_cast<uint4*>(&Blds[r * 64 + cg]) =
          *reinterpret_cast<const uint4*>(wq + (size_t)(nbase + r) * DIMC + ks + cg);
    }
    __syncthreads();

    for (int kk = 0; kk < 64; kk += 32) {
      FragBF a;
      for (int i = 0; i < 8; ++i) {
        a.u[i]     = Alds[(wave * 16 + l15) * 64 + kk + c0 + i];
        a.u[8 + i] = Alds[(wave * 16 + l15) * 64 + kk + c0 + 16 + i];
      }
      for (int nt = 0; nt < 4; ++nt) {
        FragBF bfr;
        for (int i = 0; i < 8; ++i) {
          bfr.u[i]     = Blds[(nt * 16 + l15) * 64 + kk + c0 + i];
          bfr.u[8 + i] = Blds[(nt * 16 + l15) * 64 + kk + c0 + 16 + i];
        }
        acc[nt].v = __builtin_amdgcn_wmma_f32_16x16x32_bf16(
            false, a.v, false, bfr.v, (short)0, acc[nt].v, false, false);
      }
    }
    __syncthreads();
  }

  // scatter-store into q/k/v bf16 buffers (q pre-scaled by 0.125)
  for (int nt = 0; nt < 4; ++nt) {
    int J     = nbase + nt * 16 + l15;
    int which = J / DIMC;
    int jr    = J - which * DIMC;
    int head  = jr >> 6;
    int c     = jr & 63;
    float bias = qkv_b[J];
    unsigned short* outp = (which == 0) ? qout : ((which == 1) ? kout : vout);
    float scl = (which == 0) ? 0.125f : 1.0f;
    for (int r = 0; r < 8; ++r) {
      int M = r + ((lane >> 4) << 3);
      int R = mbase + wave * 16 + M;
      if (R < ROWS) {
        int w = R / NTOK, t = R - w * NTOK;
        outp[((size_t)(w * NHEADS + head) * NTOK + t) * HD + c] =
            f2bf((acc[nt].f[r] + bias) * scl);
      }
    }
  }
}

// =====================================================================
// Kernel 2: per-(window,head) attention.
//   S (bf16 in LDS) = Qs K^T; + decomposed rel-pos bias; softmax with
//   1/rowsum deferred to the PV epilogue; O = P V via bf16 WMMA.
// =====================================================================
__global__ __launch_bounds__(256) void attn_kernel(
    const unsigned short* __restrict__ qb, const unsigned short* __restrict__ kb,
    const unsigned short* __restrict__ vb,
    const float* __restrict__ rel_pos_h, const float* __restrict__ rel_pos_w,
    unsigned short* __restrict__ obuf)
{
  __shared__ __align__(16) unsigned short Qs[208 * 64];   // scaled bf16 Q
  __shared__ __align__(16) unsigned short Ks[208 * 64];
  __shared__ __align__(16) unsigned short Vt[64 * 224];   // V transposed
  __shared__ __align__(16) unsigned short S[208 * 224];   // scores -> exp (bf16)
  __shared__ float relh[196 * 14];
  __shared__ float relw[196 * 14];
  __shared__ float invsum[208];

  const int tid  = threadIdx.x;
  const int lane = tid & 31;
  const int wave = tid >> 5;
  const int c0   = (lane >> 4) * 8;
  const int l15  = lane & 15;
  const int bh   = blockIdx.x;
  const int wwin = bh / NHEADS;
  const int head = bh - wwin * NHEADS;

  const unsigned short* qsrc = qb + (size_t)bh * NTOK * HD;
  const unsigned short* ksrc = kb + (size_t)bh * NTOK * HD;
  const unsigned short* vsrc = vb + (size_t)bh * NTOK * HD;

  // ---- stage Q, K (vector copies, zero-padded rows) ----
  for (int i = tid; i < 208 * 8; i += 256) {
    int t = i >> 3, cg = (i & 7) * 8;
    uint4 val = make_uint4(0u, 0u, 0u, 0u);
    if (t < NTOK) val = *reinterpret_cast<const uint4*>(qsrc + t * 64 + cg);
    *reinterpret_cast<uint4*>(&Qs[t * 64 + cg]) = val;
  }
  for (int i = tid; i < 208 * 8; i += 256) {
    int t = i >> 3, cg = (i & 7) * 8;
    uint4 val = make_uint4(0u, 0u, 0u, 0u);
    if (t < NTOK) val = *reinterpret_cast<const uint4*>(ksrc + t * 64 + cg);
    *reinterpret_cast<uint4*>(&Ks[t * 64 + cg]) = val;
  }
  // ---- V transpose into Vt[c][m], zero pad cols >=196 ----
  for (int i = tid; i < NTOK * 64; i += 256) {
    int m = i >> 6, c = i & 63;
    Vt[c * 224 + m] = vsrc[i];
  }
  for (int i = tid; i < 64 * 28; i += 256) {
    int c = i / 28, m = NTOK + (i - c * 28);
    Vt[c * 224 + m] = 0;
  }
  // ---- zero S pad rows (196..207) and pad cols (208..223) ----
  for (int i = tid; i < 12 * 224; i += 256) S[196 * 224 + i] = 0;
  for (int i = tid; i < NTOK * 16; i += 256) {
    int n = i >> 4;
    S[n * 224 + 208 + (i & 15)] = 0;
  }
  __syncthreads();

  // ---- S = Qs K^T : 13x13 tiles, K-dim 64; store bf16 ----
  for (int tI = wave; tI < 169; tI += 8) {
    int mt = tI / 13, ntt = tI - mt * 13;
    FragF accv;
    for (int r = 0; r < 8; ++r) accv.f[r] = 0.f;
    for (int kk = 0; kk < 64; kk += 32) {
      FragBF a, b;
      for (int i = 0; i < 8; ++i) {
        a.u[i]     = Qs[(mt * 16 + l15) * 64 + kk + c0 + i];
        a.u[8 + i] = Qs[(mt * 16 + l15) * 64 + kk + c0 + 16 + i];
        b.u[i]     = Ks[(ntt * 16 + l15) * 64 + kk + c0 + i];
        b.u[8 + i] = Ks[(ntt * 16 + l15) * 64 + kk + c0 + 16 + i];
      }
      accv.v = __builtin_amdgcn_wmma_f32_16x16x32_bf16(
          false, a.v, false, b.v, (short)0, accv.v, false, false);
    }
    for (int r = 0; r < 8; ++r) {
      int M = mt * 16 + r + ((lane >> 4) << 3);
      int N = ntt * 16 + l15;
      if (M < NTOK) S[M * 224 + N] = f2bf(accv.f[r]);
    }
  }

  // ---- decomposed rel-pos bias tables (length-64 dots) ----
  for (int idx = tid; idx < NTOK * 14; idx += 256) {
    int n = idx / 14, kk = idx - n * 14;
    int hq = n / 14, wq = n - hq * 14;
    const float* Rh = rel_pos_h + (hq - kk + 13) * HD;
    const float* Rw = rel_pos_w + (wq - kk + 13) * HD;
    float sh = 0.f, sw = 0.f;
    for (int c = 0; c < HD; ++c) {
      float qv = bf2f(Qs[n * 64 + c]) * 8.0f;   // undo 0.125 pre-scale
      sh += qv * Rh[c];
      sw += qv * Rw[c];
    }
    relh[idx] = sh; relw[idx] = sw;
  }
  __syncthreads();

  // ---- bias + softmax numerator (exp), 1/sum deferred ----
  for (int n = tid; n < NTOK; n += 256) {
    unsigned short* Srow = &S[n * 224];
    const float* rh = &relh[n * 14];
    const float* rw = &relw[n * 14];
    float mx = -1e30f;
    for (int km = 0; km < 14; ++km) {
      float bh_ = rh[km];
      for (int lm = 0; lm < 14; ++lm) {
        int m = km * 14 + lm;
        float v = bf2f(Srow[m]) + bh_ + rw[lm];
        Srow[m] = f2bf(v);
        mx = fmaxf(mx, v);
      }
    }
    float sum = 0.f;
    for (int m = 0; m < NTOK; ++m) {
      float e = __expf(bf2f(Srow[m]) - mx);
      Srow[m] = f2bf(e);
      sum += e;
    }
    invsum[n] = 1.0f / sum;
  }
  __syncthreads();

  // ---- O = P V : 13x4 tiles, K-dim 224, all-bf16 frag loads ----
  for (int tI = wave; tI < 52; tI += 8) {
    int mt = tI >> 2, ct = tI & 3;
    FragF accv;
    for (int r = 0; r < 8; ++r) accv.f[r] = 0.f;
    for (int kc = 0; kc < 224; kc += 32) {
      FragBF a, b;
      int n = mt * 16 + l15;
      for (int i = 0; i < 8; ++i) {
        a.u[i]     = S[n * 224 + kc + c0 + i];
        a.u[8 + i] = S[n * 224 + kc + c0 + 16 + i];
        b.u[i]     = Vt[(ct * 16 + l15) * 224 + kc + c0 + i];
        b.u[8 + i] = Vt[(ct * 16 + l15) * 224 + kc + c0 + 16 + i];
      }
      accv.v = __builtin_amdgcn_wmma_f32_16x16x32_bf16(
          false, a.v, false, b.v, (short)0, accv.v, false, false);
    }
    for (int r = 0; r < 8; ++r) {
      int n = mt * 16 + r + ((lane >> 4) << 3);
      int c = ct * 16 + l15;
      if (n < NTOK)
        obuf[(size_t)(wwin * NTOK + n) * DIMC + head * HD + c] =
            f2bf(accv.f[r] * invsum[n]);
    }
  }
}

// =====================================================================
// Kernel 3: proj GEMM + bias + fused window-unpartition/crop (f32 out)
// =====================================================================
__global__ __launch_bounds__(256) void proj_gemm_kernel(
    const unsigned short* __restrict__ obuf, const unsigned short* __restrict__ wp,
    const float* __restrict__ proj_b, float* __restrict__ out)
{
  __shared__ __align__(16) unsigned short Alds[128 * 64];
  __shared__ __align__(16) unsigned short Blds[64 * 64];

  const int tid   = threadIdx.x;
  const int lane  = tid & 31;
  const int wave  = tid >> 5;
  const int mbase = blockIdx.y * 128;
  const int nbase = blockIdx.x * 64;
  const int c0    = (lane >> 4) * 8;
  const int l15   = lane & 15;

  FragF acc[4];
  for (int nt = 0; nt < 4; ++nt)
    for (int r = 0; r < 8; ++r) acc[nt].f[r] = 0.0f;

  for (int ks = 0; ks < DIMC; ks += 64) {
    for (int i = tid; i < 1024; i += 256) {
      int r = i >> 3, cg = (i & 7) * 8;
      int R = mbase + r;
      uint4 val = make_uint4(0u, 0u, 0u, 0u);
      if (R < ROWS)
        val = *reinterpret_cast<const uint4*>(obuf + (size_t)R * DIMC + ks + cg);
      *reinterpret_cast<uint4*>(&Alds[r * 64 + cg]) = val;
    }
    for (int i = tid; i < 512; i += 256) {
      int r = i >> 3, cg = (i & 7) * 8;
      *reinterpret_cast<uint4*>(&Blds[r * 64 + cg]) =
          *reinterpret_cast<const uint4*>(wp + (size_t)(nbase + r) * DIMC + ks + cg);
    }
    __syncthreads();

    for (int kk = 0; kk < 64; kk += 32) {
      FragBF a;
      for (int i = 0; i < 8; ++i) {
        a.u[i]     = Alds[(wave * 16 + l15) * 64 + kk + c0 + i];
        a.u[8 + i] = Alds[(wave * 16 + l15) * 64 + kk + c0 + 16 + i];
      }
      for (int nt = 0; nt < 4; ++nt) {
        FragBF bfr;
        for (int i = 0; i < 8; ++i) {
          bfr.u[i]     = Blds[(nt * 16 + l15) * 64 + kk + c0 + i];
          bfr.u[8 + i] = Blds[(nt * 16 + l15) * 64 + kk + c0 + 16 + i];
        }
        acc[nt].v = __builtin_amdgcn_wmma_f32_16x16x32_bf16(
            false, a.v, false, bfr.v, (short)0, acc[nt].v, false, false);
      }
    }
    __syncthreads();
  }

  for (int nt = 0; nt < 4; ++nt) {
    int J = nbase + nt * 16 + l15;
    float bias = proj_b[J];
    for (int r = 0; r < 8; ++r) {
      int M = r + ((lane >> 4) << 3);
      int R = mbase + wave * 16 + M;
      if (R < ROWS) {
        int w = R / NTOK, t = R - w * NTOK;
        int b = w / 25, wrem = w - b * 25;
        int wi = wrem / 5, wj = wrem - wi * 5;
        int ty = t / WSZ, tx = t - ty * WSZ;
        int y = wi * WSZ + ty, xx = wj * WSZ + tx;
        if (y < 64 && xx < 64)
          out[((size_t)(b * 64 + y) * 64 + xx) * DIMC + J] = acc[nt].f[r] + bias;
      }
    }
  }
}

// =====================================================================
extern "C" void kernel_launch(void* const* d_in, const int* in_sizes, int n_in,
                              void* d_out, int out_size, void* d_ws, size_t ws_size,
                              hipStream_t stream) {
  const float* x      = (const float*)d_in[0];
  const float* qkv_w  = (const float*)d_in[1];
  const float* qkv_b  = (const float*)d_in[2];
  const float* proj_w = (const float*)d_in[3];
  const float* proj_b = (const float*)d_in[4];
  const float* rposh  = (const float*)d_in[5];
  const float* rposw  = (const float*)d_in[6];
  float* out = (float*)d_out;

  const size_t Q = (size_t)BH * NTOK * HD;             // 30,105,600  (== ROWS*DIMC)
  unsigned short* qb   = (unsigned short*)d_ws;
  unsigned short* kbuf = qb + Q;
  unsigned short* vbuf = kbuf + Q;
  unsigned short* xwin = vbuf + Q;                     // aliased with obuf
  unsigned short* obuf = xwin;                         // xwin dead before attn writes
  unsigned short* wq   = xwin + Q;
  unsigned short* wp   = wq + (size_t)QKVC * DIMC;

  // pack inputs to bf16
  pack_xwin_kernel<<<ROWS, 256, 0, stream>>>(x, xwin);
  pack_w_kernel<<<(QKVC * DIMC) / 256, 256, 0, stream>>>(qkv_w, wq, QKVC * DIMC);
  pack_w_kernel<<<(DIMC * DIMC) / 256, 256, 0, stream>>>(proj_w, wp, DIMC * DIMC);

  {
    dim3 grid(QKVC / 64, (ROWS + 127) / 128);          // 36 x 307
    qkv_gemm_kernel<<<grid, 256, 0, stream>>>(xwin, wq, qkv_b, qb, kbuf, vbuf);
  }
  attn_kernel<<<BH, 256, 0, stream>>>(qb, kbuf, vbuf, rposh, rposw, obuf);
  {
    dim3 grid(DIMC / 64, (ROWS + 127) / 128);          // 12 x 307
    proj_gemm_kernel<<<grid, 256, 0, stream>>>(obuf, wp, proj_b, out);
  }
  (void)in_sizes; (void)n_in; (void)out_size; (void)ws_size;
}
